// TrunkNet_79800492360235
// MI455X (gfx1250) — compile-verified, matrix operand order checked by reading, not run on hardware
//
#include <hip/hip_runtime.h>
#include <stdint.h>

// ---------- types ----------
typedef __attribute__((ext_vector_type(16))) __bf16      v16bf;
typedef __attribute__((ext_vector_type(8)))  float        v8f;
typedef __attribute__((ext_vector_type(4)))  unsigned int u32x4;
typedef __attribute__((ext_vector_type(2)))  unsigned int u32x2;
typedef __attribute__((ext_vector_type(4)))  float        f32x4;

union Frag { v16bf v; unsigned short us[16]; u32x4 q[2]; };

// ---------- problem constants ----------
#define CPJ      8
#define HID      256
#define NTOK     (1024 * 128)
#define TPB      64            // tokens per block
#define DEC_CH   176           // 22 * CPJ
#define NSPARSE  54
#define DSTRIDE  232           // dcopy row stride (176 dec + 54 sparse + 2 pad), mult of 8
#define XSTRIDE  256
#define KIN      222           // 168 gathered + 54 sparse
#define KT0      7             // ceil(224/32) K-steps for layer 0
#define KT1      8             // 256/32 K-steps for layers 1,2
#define FRAG     512           // 16x32 bf16 fragment = 512 ushorts = 1KB
#define W0_FRAGS (16 * KT0)    // 112
#define W1_FRAGS (16 * KT1)    // 128
#define W2_FRAGS (2  * KT1)    // 16   (out padded to 32)
#define MLP_FRAGS (W0_FRAGS + W1_FRAGS + W2_FRAGS)      // 256
#define MLP_WS_ELEMS (MLP_FRAGS * FRAG)                  // 131072 ushorts / MLP

// DQ_IN node chain: [0,3,6,9,12,13,14,15,16,17,18,19,20,21] + parent walk [19,17,13,9,5,2,0]
__constant__ short c_nodes[21] = {0,3,6,9,12,13,14,15,16,17,18,19,20,21,19,17,13,9,5,2,0};

// native f32 -> bf16 convert (RNE)
__device__ __forceinline__ unsigned short f2bf(float f) {
  union { __bf16 b; unsigned short u; } c;
  c.b = (__bf16)f;
  return c.u;
}

// leaky relu, slope 0.01 < 1  =>  max(x, 0.01x)
__device__ __forceinline__ float leaky(float v) { return fmaxf(v, 0.01f * v); }

// ---------- weight repack: f32 row-major [K][N] -> bf16 WMMA B-fragments ----------
// Fragment f = nt*KT + kt, element e2 = lane*16 + e; e -> (r=e>>1, p=e&1)
// K(lane,r,p) = (r<4?0:16) + (lane<16?0:8) + 2*(r&3) + p ;  N = lane&15
__global__ void __launch_bounds__(512)
prep_weights(const float* __restrict__ w0, const float* __restrict__ w1,
             const float* __restrict__ w2, int outd, unsigned short* __restrict__ dst) {
  int f = blockIdx.x;
  const float* w; int KT, Kdim, ldw, Ndim; long base;
  if (f < W0_FRAGS)                  { w = w0; KT = KT0; Kdim = KIN; ldw = HID;  Ndim = HID;  base = 0; }
  else if (f < W0_FRAGS + W1_FRAGS)  { f -= W0_FRAGS; w = w1; KT = KT1; Kdim = HID; ldw = HID; Ndim = HID; base = (long)W0_FRAGS * FRAG; }
  else                               { f -= (W0_FRAGS + W1_FRAGS); w = w2; KT = KT1; Kdim = HID; ldw = outd; Ndim = outd; base = (long)(W0_FRAGS + W1_FRAGS) * FRAG; }
  int nt = f / KT, kt = f % KT;
  int e2 = threadIdx.x;                 // 0..511
  int lane = e2 >> 4, e = e2 & 15;
  int r = e >> 1, p = e & 1;
  int K = ((r < 4) ? 0 : 16) + ((lane < 16) ? 0 : 8) + 2 * (r & 3) + p;
  int N = lane & 15;
  int kg = kt * 32 + K, ng = nt * 16 + N;
  float v = (kg < Kdim && ng < Ndim) ? w[(long)kg * ldw + ng] : 0.0f;
  dst[base + (long)f * FRAG + e2] = f2bf(v);
}

// ---------- hidden layer: X(64xK) @ W -> H(64x256) bf16, fused bias + leaky ----------
// Wave owns one M-tile (2 waves per M-tile), loops 8 N-tiles keeping A in registers.
template <int KT>
__device__ __forceinline__ void layer_hidden(const unsigned short* __restrict__ X,
                                             const unsigned short* __restrict__ Wb,
                                             const float* __restrict__ bias,
                                             unsigned short* __restrict__ H,
                                             int lane, int wave) {
  const int lrow  = lane & 15;
  const int koff  = (lane < 16) ? 0 : 8;
  const int mhalf = (lane < 16) ? 0 : 8;
  const int mt    = wave >> 1;          // 4 M-tiles, 2 waves each
  const int nt0   = (wave & 1) * 8;     // 8 N-tiles per wave

  Frag a[KT];
  const unsigned short* arow = X + (mt * 16 + lrow) * XSTRIDE + koff;
#pragma unroll
  for (int k = 0; k < KT; ++k) {
    a[k].q[0] = *(const u32x4*)(arow + k * 32);
    a[k].q[1] = *(const u32x4*)(arow + k * 32 + 16);
  }

  unsigned short* hrow = H + (mt * 16 + mhalf) * XSTRIDE;
  for (int j = 0; j < 8; ++j) {
    const int nt = nt0 + j;
    const unsigned short* wf = Wb + (long)(nt * KT) * FRAG + lane * 16;
    v8f acc = {};
#pragma unroll
    for (int k = 0; k < KT; ++k) {
      Frag b;
      b.q[0] = *(const u32x4*)(wf + (long)k * FRAG);
      b.q[1] = *(const u32x4*)(wf + (long)k * FRAG + 8);
      acc = __builtin_amdgcn_wmma_f32_16x16x32_bf16(false, a[k].v, false, b.v,
                                                    (short)0, acc, false, false);
    }
    const int n = nt * 16 + lrow;
    const float bv = bias[n];
#pragma unroll
    for (int r = 0; r < 8; ++r)
      hrow[r * XSTRIDE + n] = f2bf(leaky(acc[r] + bv));
  }
}

// ---------- output layer: X(64x256) @ W2(256x32pad); scatter or emit ----------
// mode 0: write node-9 channels (72+n) into dcopy
// mode 1: write nodes {6,9,12} channels into dcopy
// mode 2: write f32 result to global out [tok][24]
__device__ __forceinline__ void layer_out(const unsigned short* __restrict__ X,
                                          const unsigned short* __restrict__ Wb,
                                          const float* __restrict__ bias,
                                          unsigned short* __restrict__ dcopy,
                                          float* __restrict__ out, int tok0,
                                          int outd, int mode, int lane, int wave) {
  const int lrow  = lane & 15;
  const int koff  = (lane < 16) ? 0 : 8;
  const int mhalf = (lane < 16) ? 0 : 8;
  const int mt = wave >> 1, nt = wave & 1;   // 4 M-tiles x 2 N-tiles = 8 tiles = 8 waves
  const unsigned short* arow = X + (mt * 16 + lrow) * XSTRIDE + koff;
  const unsigned short* wf = Wb + (long)(nt * KT1) * FRAG + lane * 16;
  v8f acc = {};
#pragma unroll
  for (int k = 0; k < KT1; ++k) {
    Frag a, b;
    a.q[0] = *(const u32x4*)(arow + k * 32);
    a.q[1] = *(const u32x4*)(arow + k * 32 + 16);
    b.q[0] = *(const u32x4*)(wf + (long)k * FRAG);
    b.q[1] = *(const u32x4*)(wf + (long)k * FRAG + 8);
    acc = __builtin_amdgcn_wmma_f32_16x16x32_bf16(false, a.v, false, b.v,
                                                  (short)0, acc, false, false);
  }
  const int n = nt * 16 + lrow;
  if (n < outd) {
    const float bv = bias[n];
    const int ch = (mode == 0) ? (72 + n)
                 : ((n < 8) ? (48 + n) : (n < 16) ? (64 + n) : (80 + n));
    const int row0 = mt * 16 + mhalf;
#pragma unroll
    for (int r = 0; r < 8; ++r) {
      const float v = acc[r] + bv;
      if (mode == 2) out[(size_t)(tok0 + row0 + r) * 24 + n] = v;
      else           dcopy[(row0 + r) * DSTRIDE + ch] = f2bf(v);
    }
  }
}

// ---------- gather (u32 pairs): x[0..167]=dcopy[DQ_IN], x[168..221]=sparse, x[222..223]=0
// channel pairs (2i,2i+1) always land in the same 8-aligned node group -> contiguous & aligned
__device__ __forceinline__ void gather_x(const unsigned short* __restrict__ dcopy,
                                         unsigned short* __restrict__ X, int tid) {
  for (int i = tid; i < TPB * 112; i += 256) {
    int tok = i / 112, c = (i % 112) * 2;
    unsigned int v;
    if (c < 168)      v = *(const unsigned int*)(dcopy + tok * DSTRIDE + c_nodes[c >> 3] * CPJ + (c & 7));
    else if (c < 222) v = *(const unsigned int*)(dcopy + tok * DSTRIDE + DEC_CH + (c - 168));
    else              v = 0u;
    *(unsigned int*)(X + tok * XSTRIDE + c) = v;
  }
}

extern __shared__ unsigned short smem[];

__global__ void __launch_bounds__(256)
trunk_kernel(const float* __restrict__ sp, const float* __restrict__ dec,
             const unsigned short* __restrict__ wA, const float* __restrict__ b0a,
             const float* __restrict__ b1a, const float* __restrict__ b2a,
             const unsigned short* __restrict__ wB, const float* __restrict__ b0b,
             const float* __restrict__ b1b, const float* __restrict__ b2b,
             const unsigned short* __restrict__ wC, const float* __restrict__ b0c,
             const float* __restrict__ b1c, const float* __restrict__ b2c,
             float* __restrict__ out) {
  unsigned short* dcopy = smem;                         // 64 x 232
  unsigned short* bufA  = smem + TPB * DSTRIDE;         // 64 x 256 (x / h2)
  unsigned short* bufB  = bufA + TPB * XSTRIDE;         // 64 x 256 (h)
  const int tid  = threadIdx.x;
  const int lane = tid & 31, wave = tid >> 5;
  const int tok0 = blockIdx.x * TPB;

  // stage decoder channels (vectorized: 44 float4 per token) + sparse into LDS (bf16)
  for (int i = tid; i < TPB * (DEC_CH / 4); i += 256) {
    int tok = i / (DEC_CH / 4), c4 = (i % (DEC_CH / 4)) * 4;
    f32x4 d4 = *(const f32x4*)(dec + (size_t)(tok0 + tok) * DEC_CH + c4);
    u32x2 p;
    p[0] = (unsigned int)f2bf(d4[0]) | ((unsigned int)f2bf(d4[1]) << 16);
    p[1] = (unsigned int)f2bf(d4[2]) | ((unsigned int)f2bf(d4[3]) << 16);
    *(u32x2*)(dcopy + tok * DSTRIDE + c4) = p;
  }
  for (int i = tid; i < TPB * (NSPARSE / 2); i += 256) {
    int tok = i / (NSPARSE / 2), c2 = (i % (NSPARSE / 2)) * 2;
    float s0 = sp[(size_t)(tok0 + tok) * NSPARSE + c2];
    float s1 = sp[(size_t)(tok0 + tok) * NSPARSE + c2 + 1];
    *(unsigned int*)(dcopy + tok * DSTRIDE + DEC_CH + c2) =
        (unsigned int)f2bf(s0) | ((unsigned int)f2bf(s1) << 16);
  }
  __syncthreads();

#pragma unroll 1
  for (int m = 0; m < 3; ++m) {
    const unsigned short* wb = (m == 0) ? wA : (m == 1) ? wB : wC;
    const float* b0 = (m == 0) ? b0a : (m == 1) ? b0b : b0c;
    const float* b1 = (m == 0) ? b1a : (m == 1) ? b1b : b1c;
    const float* b2 = (m == 0) ? b2a : (m == 1) ? b2b : b2c;

    gather_x(dcopy, bufA, tid);
    __syncthreads();
    layer_hidden<KT0>(bufA, wb, b0, bufB, lane, wave);
    __syncthreads();
    layer_hidden<KT1>(bufB, wb + (long)W0_FRAGS * FRAG, b1, bufA, lane, wave);
    __syncthreads();
    layer_out(bufA, wb + (long)(W0_FRAGS + W1_FRAGS) * FRAG, b2, dcopy, out, tok0,
              (m == 0) ? 8 : 24, m, lane, wave);
    __syncthreads();
  }
}

extern "C" void kernel_launch(void* const* d_in, const int* in_sizes, int n_in,
                              void* d_out, int out_size, void* d_ws, size_t ws_size,
                              hipStream_t stream) {
  (void)in_sizes; (void)n_in; (void)out_size; (void)ws_size;
  const float* sp   = (const float*)d_in[0];
  const float* dec  = (const float*)d_in[1];
  const float* w0_1 = (const float*)d_in[2];
  const float* b0_1 = (const float*)d_in[3];
  const float* w1_1 = (const float*)d_in[4];
  const float* b1_1 = (const float*)d_in[5];
  const float* w2_1 = (const float*)d_in[6];
  const float* b2_1 = (const float*)d_in[7];
  const float* w0_2 = (const float*)d_in[8];
  const float* b0_2 = (const float*)d_in[9];
  const float* w1_2 = (const float*)d_in[10];
  const float* b1_2 = (const float*)d_in[11];
  const float* w2_2 = (const float*)d_in[12];
  const float* b2_2 = (const float*)d_in[13];
  const float* w0_3 = (const float*)d_in[14];
  const float* b0_3 = (const float*)d_in[15];
  const float* w1_3 = (const float*)d_in[16];
  const float* b1_3 = (const float*)d_in[17];
  const float* w2_3 = (const float*)d_in[18];
  const float* b2_3 = (const float*)d_in[19];

  unsigned short* ws  = (unsigned short*)d_ws;
  unsigned short* wsA = ws;
  unsigned short* wsB = ws + (size_t)MLP_WS_ELEMS;
  unsigned short* wsC = ws + (size_t)2 * MLP_WS_ELEMS;

  prep_weights<<<MLP_FRAGS, 512, 0, stream>>>(w0_1, w1_1, w2_1, 8,  wsA);
  prep_weights<<<MLP_FRAGS, 512, 0, stream>>>(w0_2, w1_2, w2_2, 24, wsB);
  prep_weights<<<MLP_FRAGS, 512, 0, stream>>>(w0_3, w1_3, w2_3, 24, wsC);

  size_t smem_bytes = (size_t)(TPB * DSTRIDE + 2 * TPB * XSTRIDE) * sizeof(unsigned short);
  trunk_kernel<<<NTOK / TPB, 256, smem_bytes, stream>>>(
      sp, dec,
      wsA, b0_1, b1_1, b2_1,
      wsB, b0_2, b1_2, b2_2,
      wsC, b0_3, b1_3, b2_3,
      (float*)d_out);
}
